// PrototypeMemoryBank_41953240547510
// MI455X (gfx1250) — matrix operand order repaired and checked by reading, not compiled, not run on hardware
//
#include <hip/hip_runtime.h>
#include <hip/hip_bf16.h>
#include <math.h>

// ---------------------------------------------------------------------------
// Problem constants (from reference): B=2048, M=100000, D=256, K=16
// ---------------------------------------------------------------------------
#define BQ       2048
#define MP       100000
#define DC       256
#define TOPK     16
#define KSTEPS   8          // 256 / 32 (WMMA K per step)
#define NSLICES  50         // M split into 50 disjoint slices
#define PT_PER_SLICE 125    // 125 tiles * 16 protos = 2000 protos / slice
#define NQT      128        // 2048 / 16 query tiles

typedef _Float16 v8h  __attribute__((ext_vector_type(8)));
typedef _Float16 v16h __attribute__((ext_vector_type(16)));
typedef float    v8f  __attribute__((ext_vector_type(8)));
typedef float    v4f  __attribute__((ext_vector_type(4)));

union U16H { v16h v; v8h h[2]; };

// ---------------------------------------------------------------------------
// Kernel 1: row-wise L2 normalize fp32 -> f16 (one wave per row)
// ---------------------------------------------------------------------------
__global__ __launch_bounds__(256)
void pmb_normalize_f16(const float* __restrict__ src, _Float16* __restrict__ dst, int rows) {
    const int wave = (int)(blockIdx.x * (blockDim.x >> 5)) + ((int)threadIdx.x >> 5);
    const int lane = (int)threadIdx.x & 31;
    if (wave >= rows) return;

    const float4* p = (const float4*)(src + (size_t)wave * DC);
    float4 a = p[lane * 2 + 0];
    float4 b = p[lane * 2 + 1];

    float ss = a.x*a.x + a.y*a.y + a.z*a.z + a.w*a.w
             + b.x*b.x + b.y*b.y + b.z*b.z + b.w*b.w;
    #pragma unroll
    for (int off = 16; off > 0; off >>= 1)
        ss += __shfl_xor(ss, off, 32);

    const float n = sqrtf(ss);
    const float s = 1.0f / fmaxf(n, 1e-8f);

    v8h o;
    o[0] = (_Float16)(a.x * s); o[1] = (_Float16)(a.y * s);
    o[2] = (_Float16)(a.z * s); o[3] = (_Float16)(a.w * s);
    o[4] = (_Float16)(b.x * s); o[5] = (_Float16)(b.y * s);
    o[6] = (_Float16)(b.z * s); o[7] = (_Float16)(b.w * s);
    *(v8h*)(dst + (size_t)wave * DC + lane * 8) = o;
}

// ---------------------------------------------------------------------------
// Kernel 2: fused cosine-sim GEMM (WMMA f16 -> f32) + per-slice exact top-16
// grid = (128 query tiles, 50 M-slices), block = 32 threads (one wave)
// ---------------------------------------------------------------------------
__global__ __launch_bounds__(32)
void pmb_sim_topk(const _Float16* __restrict__ qn,
                  const _Float16* __restrict__ pn,
                  float* __restrict__ pvals,
                  int*   __restrict__ pidxs) {
    __shared__ __align__(16) float tile[16 * 16];

    const int qt   = (int)blockIdx.x;
    const int sl   = (int)blockIdx.y;
    const int lane = (int)threadIdx.x;
    const int half = lane >> 4;      // 0: lanes 0-15, 1: lanes 16-31
    const int lr   = lane & 15;

    // ---- Hoist A (16 queries x 256) into registers, WMMA 16x32 f16 layout:
    //      lane<16 : K {k0..k0+7} then {k0+16..k0+23}
    //      lane>=16: K {k0+8..k0+15} then {k0+24..k0+31}
    const _Float16* qrow = qn + (size_t)(qt * 16 + lr) * DC;
    v16h A[KSTEPS];
    #pragma unroll
    for (int k = 0; k < KSTEPS; ++k) {
        const int k0 = k * 32;
        U16H u;
        u.h[0] = *(const v8h*)(qrow + k0 + half * 8);
        u.h[1] = *(const v8h*)(qrow + k0 + 16 + half * 8);
        A[k] = u.v;
    }

    // ---- running sorted (descending) top-16 per owner lane (lanes duplicated
    //      across halves so the wave stays convergent)
    float vals[TOPK]; int idxs[TOPK];
    #pragma unroll
    for (int s = 0; s < TOPK; ++s) { vals[s] = -3.4e38f; idxs[s] = 0; }

    const int pbase0 = sl * PT_PER_SLICE * 16;

    for (int pt = 0; pt < PT_PER_SLICE; ++pt) {
        const int pbase = pbase0 + pt * 16;

        // B layout (32x16 f16): lane = column (proto), lanes 0-15 K{k0..k0+7,k0+16..k0+23},
        // lanes 16-31 K{k0+8..k0+15,k0+24..k0+31} -> two 16B reads per lane.
        const _Float16* prow = pn + (size_t)(pbase + lr) * DC;

        // pull next tile's B row toward the WGP while this tile computes
        // (separate scheduling region, does not perturb the group counts below)
        if (pt + 1 < PT_PER_SLICE) {
            const _Float16* nrow = prow + 16 * DC;
            __builtin_prefetch(nrow,       0, 3);
            __builtin_prefetch(nrow +  64, 0, 3);
            __builtin_prefetch(nrow + 128, 0, 3);
            __builtin_prefetch(nrow + 192, 0, 3);
        }

        // ---- 3-deep software pipeline on B so WMMAs overlap the loads
        U16H b[3];
        #define PMB_LOADB(dst, kk) {                                          \
            const int _k0 = (kk) * 32;                                        \
            (dst).h[0] = *(const v8h*)(prow + _k0 + half * 8);                \
            (dst).h[1] = *(const v8h*)(prow + _k0 + 16 + half * 8);           \
        }
        PMB_LOADB(b[0], 0);
        PMB_LOADB(b[1], 1);

        v8f c = {0.f, 0.f, 0.f, 0.f, 0.f, 0.f, 0.f, 0.f};
        #pragma unroll
        for (int k = 0; k < KSTEPS; ++k) {
            if (k + 2 < KSTEPS) PMB_LOADB(b[(k + 2) % 3], k + 2);
            c = __builtin_amdgcn_wmma_f32_16x16x32_f16(
                    false, A[k], false, b[k % 3].v, (short)0, c, false, false);
        }
        #undef PMB_LOADB

        // ---- pin the pipelined schedule:
        //   6 loads (pairs k0..k2), then (1 WMMA, 2 loads) x5, then 3 WMMAs.
        // mask 0x020 = VMEM read, 0x008 = MFMA/WMMA group.
        __builtin_amdgcn_sched_group_barrier(0x020, 6, 0);
        #pragma unroll
        for (int g = 0; g < 5; ++g) {
            __builtin_amdgcn_sched_group_barrier(0x008, 1, 0);
            __builtin_amdgcn_sched_group_barrier(0x020, 2, 0);
        }
        __builtin_amdgcn_sched_group_barrier(0x008, 3, 0);

        // ---- transpose scores to owner lanes via LDS
        // C layout: lane l, vgpr i -> row (half? i+8 : i), col (l&15)
        #pragma unroll
        for (int i = 0; i < 8; ++i) {
            const int row = half ? (i + 8) : i;
            tile[row * 16 + lr] = c[i];
        }
        __syncthreads();   // single-wave WG: barrier is a NOP, acts as mem fence

        const v4f* tr = (const v4f*)(tile + lr * 16);
        v4f t0 = tr[0], t1 = tr[1], t2 = tr[2], t3 = tr[3];
        __syncthreads();

        float cand[16];
        cand[ 0]=t0[0]; cand[ 1]=t0[1]; cand[ 2]=t0[2]; cand[ 3]=t0[3];
        cand[ 4]=t1[0]; cand[ 5]=t1[1]; cand[ 6]=t1[2]; cand[ 7]=t1[3];
        cand[ 8]=t2[0]; cand[ 9]=t2[1]; cand[10]=t2[2]; cand[11]=t2[3];
        cand[12]=t3[0]; cand[13]=t3[1]; cand[14]=t3[2]; cand[15]=t3[3];

        // fast path: skip whole tile unless it can change the top-16
        float m = cand[0];
        #pragma unroll
        for (int j = 1; j < 16; ++j) m = fmaxf(m, cand[j]);

        if (m > vals[TOPK - 1]) {
            #pragma unroll
            for (int j = 0; j < 16; ++j) {
                float cv = cand[j];
                if (cv > vals[TOPK - 1]) {
                    int ci = pbase + j;
                    #pragma unroll
                    for (int s = 0; s < TOPK; ++s) {
                        const bool gt = cv > vals[s];
                        const float tv = vals[s]; const int ti = idxs[s];
                        vals[s] = gt ? cv : tv;  idxs[s] = gt ? ci : ti;
                        cv      = gt ? tv : cv;  ci      = gt ? ti : ci;
                    }
                }
            }
        }
    }

    // ---- emit per-slice sorted top-16 (lanes 0..15 own rows 0..15)
    if (half == 0) {
        const size_t base = (((size_t)qt * NSLICES + sl) * 16 + lr) * TOPK;
        float* vo = pvals + base;
        int*   io = pidxs + base;
        #pragma unroll
        for (int s = 0; s < TOPK; ++s) { vo[s] = vals[s]; io[s] = idxs[s]; }
    }
}

// ---------------------------------------------------------------------------
// Kernel 3: merge 50 sorted per-slice top-16 lists -> global top-16,
//           then gather the 16 prototype rows and average. 1 block / query.
// ---------------------------------------------------------------------------
__global__ __launch_bounds__(256)
void pmb_merge_gather(const float* __restrict__ pvals,
                      const int*   __restrict__ pidxs,
                      const float* __restrict__ protos,
                      float* __restrict__ out) {
    __shared__ int topIdx[TOPK];

    const int q  = (int)blockIdx.x;
    const int qt = q >> 4;
    const int r  = q & 15;

    if (threadIdx.x == 0) {
        float vals[TOPK]; int idxs[TOPK];
        #pragma unroll
        for (int s = 0; s < TOPK; ++s) { vals[s] = -3.4e38f; idxs[s] = 0; }

        for (int sl = 0; sl < NSLICES; ++sl) {
            const size_t base = (((size_t)qt * NSLICES + sl) * 16 + r) * TOPK;
            const float* v  = pvals + base;
            const int*   ii = pidxs + base;
            for (int j = 0; j < TOPK; ++j) {
                float cv = v[j];
                if (cv <= vals[TOPK - 1]) break;   // slice list is sorted desc
                int ci = ii[j];
                #pragma unroll
                for (int s = 0; s < TOPK; ++s) {
                    const bool gt = cv > vals[s];
                    const float tv = vals[s]; const int ti = idxs[s];
                    vals[s] = gt ? cv : tv;  idxs[s] = gt ? ci : ti;
                    cv      = gt ? tv : cv;  ci      = gt ? ti : ci;
                }
            }
        }
        #pragma unroll
        for (int s = 0; s < TOPK; ++s) topIdx[s] = idxs[s];
    }
    __syncthreads();

    const int d = (int)threadIdx.x;     // 256 threads = 256 dims
    float sum = 0.f;
    #pragma unroll
    for (int t = 0; t < TOPK; ++t)
        sum += protos[(size_t)topIdx[t] * DC + d];
    out[(size_t)q * DC + d] = sum * (1.0f / (float)TOPK);
}

// ---------------------------------------------------------------------------
// Launcher
// ---------------------------------------------------------------------------
extern "C" void kernel_launch(void* const* d_in, const int* in_sizes, int n_in,
                              void* d_out, int out_size, void* d_ws, size_t ws_size,
                              hipStream_t stream) {
    const float* query  = (const float*)d_in[0];   // [2048, 256] f32
    const float* protos = (const float*)d_in[1];   // [100000, 256] f32
    float* out = (float*)d_out;                    // [2048, 256] f32
    (void)in_sizes; (void)n_in; (void)out_size; (void)ws_size;

    // workspace carve (all 16B aligned)
    char* ws = (char*)d_ws;
    _Float16* qn = (_Float16*)ws;  ws += (size_t)BQ * DC * sizeof(_Float16);   //  1.0 MB
    _Float16* pn = (_Float16*)ws;  ws += (size_t)MP * DC * sizeof(_Float16);   // 51.2 MB
    float* pvals = (float*)ws;     ws += (size_t)NQT * NSLICES * 16 * TOPK * sizeof(float); // 6.55 MB
    int*   pidxs = (int*)ws;                                                   // 6.55 MB

    // 1) normalize to f16 (8 waves / block, one wave per row)
    pmb_normalize_f16<<<BQ / 8, 256, 0, stream>>>(query, qn, BQ);
    pmb_normalize_f16<<<MP / 8, 256, 0, stream>>>(protos, pn, MP);

    // 2) WMMA cosine-sim + per-slice top-16
    dim3 grid(NQT, NSLICES);
    pmb_sim_topk<<<grid, 32, 0, stream>>>(qn, pn, pvals, pidxs);

    // 3) merge slices + gather-mean
    pmb_merge_gather<<<BQ, 256, 0, stream>>>(pvals, pidxs, protos, out);
}
